// GIN_X_BN_77558519431976
// MI455X (gfx1250) — matrix operand order repaired and checked by reading, not compiled, not run on hardware
//
#include <hip/hip_runtime.h>

#define N_NODES 100000
#define N_EDGES 1600000
#define D_FEAT  128          // aggregation feature width for all 3 layers
#define D_OUT   64
#define BN_EPS  1e-5f
#define MT_PER_BLOCK 10      // 6250 m-tiles = 625 blocks x 10

typedef __attribute__((ext_vector_type(16))) __bf16 v16bf;
typedef __attribute__((ext_vector_type(8)))  float  v8f;

// ---- CDNA5 async global->LDS copy (ASYNCcnt) -------------------------------
// LDS destination address is taken from the vdst VGPR; low 32 bits of a generic pointer to __shared__
// are the wave-relative LDS byte offset (aperture upper bits discarded).
__device__ __forceinline__ void async_copy_b128(unsigned lds_off, const void* sbase,
                                                unsigned voff) {
    asm volatile("global_load_async_to_lds_b128 %0, %1, %2"
                 :: "v"(lds_off), "v"(voff), "s"(sbase) : "memory");
}
__device__ __forceinline__ void wait_async0() {
    asm volatile("s_wait_asynccnt 0" ::: "memory");
}

// ---------------- agg init: agg = (1+eps)*x with eps=0 -> copy ----------------
__global__ void copy_f4(const float* __restrict__ in, float* __restrict__ out, int n4) {
    int i = blockIdx.x * blockDim.x + threadIdx.x;
    if (i < n4) ((float4*)out)[i] = ((const float4*)in)[i];
}

// ---------------- edge scatter: agg[dst] += x[src], one wave per edge ----------------
// 1.6M edges * 128 f32 -> 204.8M global_atomic_add_f32 per layer, all L2-resident
// (feature matrix = 51.2 MB << 192 MB L2). This is the roofline-critical kernel.
__global__ void edge_scatter(const float* __restrict__ x, float* __restrict__ agg,
                             const int* __restrict__ src, const int* __restrict__ dst) {
    int gid = blockIdx.x * blockDim.x + threadIdx.x;
    int e = gid >> 5;                 // edge id (wave per edge)
    if (e >= N_EDGES) return;
    int g = (gid & 31) << 2;          // feature offset 0..124, float4 per lane
    if ((gid & 31) == 0 && e + 4096 < N_EDGES) {
        __builtin_prefetch(src + e + 4096, 0, 0);   // global_prefetch_b8
        __builtin_prefetch(dst + e + 4096, 0, 0);
    }
    int s = src[e], d = dst[e];
    float4 v = *(const float4*)(x + (size_t)s * D_FEAT + g);
    float* p = agg + (size_t)d * D_FEAT + g;
    unsafeAtomicAdd(p + 0, v.x);      // global_atomic_add_f32 at L2 atomic units
    unsafeAtomicAdd(p + 1, v.y);
    unsafeAtomicAdd(p + 2, v.z);
    unsafeAtomicAdd(p + 3, v.w);
}

// ---------------- fused GEMM + bias (+ReLU) via v_wmma_f32_16x16x32_bf16 ----------------
// Block = NOUT/16 waves; wave w owns column-tile w. Whole W (<=64 KB) is staged
// into LDS once per block via async b128 copies, converted to bf16 B-fragments
// held in registers, then the block sweeps MT_PER_BLOCK m-tiles: inner loop is
// contiguous A loads + 4 WMMAs per tile.
template<int NOUT, bool RELU>
__global__ void gemm_wmma(const float* __restrict__ A, const float* __restrict__ W,
                          const float* __restrict__ bias, float* __restrict__ H) {
    __shared__ float sW[128 * NOUT];

    const int tid  = threadIdx.x;
    const int lane = tid & 31;
    const int nt   = tid >> 5;              // wave id == n-tile id
    const int lo   = lane & 15;
    const int hi   = lane >> 4;             // half-wave select per ISA layout
    const int n    = nt * 16 + lo;

    // ---- stage W into LDS: 32*NOUT b128 chunks, 16 per thread (ASYNCcnt) ----
    {
        unsigned lds_base = (unsigned)(size_t)(void*)sW;   // wave-relative LDS offset
        #pragma unroll
        for (int it = 0; it < 16; ++it) {
            unsigned byte_off = (unsigned)(tid + it * (NOUT * 16)) * 16u;
            async_copy_b128(lds_base + byte_off, W, byte_off);
        }
        wait_async0();
        __syncthreads();
    }

    // ---- build B fragments (32x16 bf16 per k-step) once per wave from LDS ----
    // elems 0..7 = K k0+8*hi+e, elems 8..15 = K k0+16+8*hi+(e-8), col = lo
    v16bf bfrag[4];
    #pragma unroll
    for (int k = 0; k < 4; ++k) {
        const float* wcol = sW + (size_t)(k * 32 + hi * 8) * NOUT + n;
        #pragma unroll
        for (int e = 0; e < 16; ++e) bfrag[k][e] = (__bf16)wcol[(size_t)(e + (e & 8)) * NOUT];
    }

    const float bv = bias[n];

    // ---- sweep m-tiles, reusing registered B fragments ----
    const int mt0 = blockIdx.x * MT_PER_BLOCK;
    for (int mt = mt0; mt < mt0 + MT_PER_BLOCK; ++mt) {
        v8f c = {0.f, 0.f, 0.f, 0.f, 0.f, 0.f, 0.f, 0.f};
        // A fragment: lane row m = lo; elems 0..7 = K k0+8*hi+e, elems 8..15 = +16
        const float* arow = A + (size_t)(mt * 16 + lo) * 128 + hi * 8;
        #pragma unroll
        for (int k = 0; k < 4; ++k) {
            v16bf a;
            #pragma unroll
            for (int e = 0; e < 8; ++e) a[e]     = (__bf16)arow[k * 32 + e];
            #pragma unroll
            for (int e = 0; e < 8; ++e) a[8 + e] = (__bf16)arow[k * 32 + 16 + e];
            c = __builtin_amdgcn_wmma_f32_16x16x32_bf16(
                    /*neg_a=*/false, a, /*neg_b=*/false, bfrag[k],
                    /*c_mod=*/(short)0, c, /*reuse_a=*/false, /*reuse_b=*/false);
        }
        // C/D layout: VGPR r -> row r + 8*hi, col = lo
        #pragma unroll
        for (int r = 0; r < 8; ++r) {
            float v = c[r] + bv;
            if (RELU) v = fmaxf(v, 0.f);
            H[(size_t)(mt * 16 + hi * 8 + r) * NOUT + n] = v;
        }
    }
}

// ---------------- BatchNorm over nodes ----------------
__global__ void zero_buf(float* p, int n) {
    int i = blockIdx.x * blockDim.x + threadIdx.x;
    if (i < n) p[i] = 0.f;
}

__global__ void bn_stats(const float* __restrict__ h, float* __restrict__ stats) {
    __shared__ float ssum[D_OUT], ssq[D_OUT];
    int t = threadIdx.x;                      // 256 threads
    if (t < D_OUT) { ssum[t] = 0.f; ssq[t] = 0.f; }
    __syncthreads();
    int col = t & 63;
    float ls = 0.f, lq = 0.f;
    for (int r = blockIdx.x * 4 + (t >> 6); r < N_NODES; r += gridDim.x * 4) {
        float v = h[(size_t)r * D_OUT + col];
        ls += v; lq += v * v;
    }
    atomicAdd(&ssum[col], ls);                // ds_add_f32
    atomicAdd(&ssq[col],  lq);
    __syncthreads();
    if (t < D_OUT) {
        unsafeAtomicAdd(&stats[t],          ssum[t]);
        unsafeAtomicAdd(&stats[D_OUT + t],  ssq[t]);
    }
}

__global__ void bn_apply(float* __restrict__ h, const float* __restrict__ stats,
                         const float* __restrict__ gamma, const float* __restrict__ beta) {
    int i = blockIdx.x * blockDim.x + threadIdx.x;
    if (i >= N_NODES * D_OUT) return;
    int col = i & 63;
    const float inv_n = 1.f / (float)N_NODES;
    float mean = stats[col] * inv_n;
    float var  = stats[D_OUT + col] * inv_n - mean * mean;
    h[i] = (h[i] - mean) * rsqrtf(var + BN_EPS) * gamma[col] + beta[col];
}

extern "C" void kernel_launch(void* const* d_in, const int* in_sizes, int n_in,
                              void* d_out, int out_size, void* d_ws, size_t ws_size,
                              hipStream_t stream) {
    const float* x     = (const float*)d_in[0];
    const int*   ei    = (const int*)d_in[1];
    const int*   src   = ei;                     // edge_index row 0
    const int*   dst   = ei + N_EDGES;           // edge_index row 1
    const float* W1    = (const float*)d_in[2];
    const float* b1    = (const float*)d_in[3];
    const float* Wm    = (const float*)d_in[4];
    const float* bm    = (const float*)d_in[5];
    const float* W2    = (const float*)d_in[6];
    const float* b2    = (const float*)d_in[7];
    const float* gamma = (const float*)d_in[8];
    const float* beta  = (const float*)d_in[9];
    float* out = (float*)d_out;

    // workspace: [256 floats BN stats][N*128 bufA][N*128 bufH]  (~102.4 MB)
    float* stats = (float*)d_ws;
    float* bufA  = stats + 256;
    float* bufH  = bufA + (size_t)N_NODES * D_FEAT;

    const int NF4    = N_NODES * D_FEAT / 4;
    const int COPY_B = (NF4 + 255) / 256;
    const int EDGE_B = (N_EDGES * 32 + 255) / 256;
    const int GEMM_B = (N_NODES / 16) / MT_PER_BLOCK;   // 625

    // layer 1: agg = x + scatter(x); h = relu(agg @ W1 + b1)
    copy_f4     <<<COPY_B, 256, 0, stream>>>(x, bufA, NF4);
    edge_scatter<<<EDGE_B, 256, 0, stream>>>(x, bufA, src, dst);
    gemm_wmma<128, true><<<GEMM_B, 256, 0, stream>>>(bufA, W1, b1, bufH);

    // layer 2 (middle): agg = h + scatter(h); h = relu(agg @ Wm + bm)
    copy_f4     <<<COPY_B, 256, 0, stream>>>(bufH, bufA, NF4);
    edge_scatter<<<EDGE_B, 256, 0, stream>>>(bufH, bufA, src, dst);
    gemm_wmma<128, true><<<GEMM_B, 256, 0, stream>>>(bufA, Wm, bm, bufH);

    // layer 3: agg = h + scatter(h); out_pre = agg @ W2 + b2  (straight to d_out)
    copy_f4     <<<COPY_B, 256, 0, stream>>>(bufH, bufA, NF4);
    edge_scatter<<<EDGE_B, 256, 0, stream>>>(bufH, bufA, src, dst);
    gemm_wmma<D_OUT, false><<<GEMM_B, 128, 0, stream>>>(bufA, W2, b2, out);

    // BatchNorm over node axis, in place on d_out
    zero_buf<<<1, 128, 0, stream>>>(stats, 2 * D_OUT);
    bn_stats<<<512, 256, 0, stream>>>(out, stats);
    bn_apply<<<(N_NODES * D_OUT + 255) / 256, 256, 0, stream>>>(out, stats, gamma, beta);
}
